// SpatialTransformer_62577673503450
// MI455X (gfx1250) — matrix operand order repaired and checked by reading, not compiled, not run on hardware
//
#include <hip/hip_runtime.h>
#include <hip/hip_bf16.h>

// ---------------- problem constants ----------------
#define BB 32
#define SS 512
#define DD 512
#define HH 8
#define DKK 64
#define DFF 2048
#define NBB 2
#define MROWS (BB * SS)   // 16384

typedef __bf16 bf16_t;
typedef __attribute__((ext_vector_type(8)))  bf16_t bf16x8;
typedef __attribute__((ext_vector_type(16))) bf16_t bf16x16;
typedef __attribute__((ext_vector_type(8)))  float  f32x8;

// ---- CDNA5 async global->LDS path (ASYNCcnt-tracked DMA) ----
#if __has_builtin(__builtin_amdgcn_global_load_async_to_lds_b128)
#define HAVE_ASYNC_LDS 1
typedef int i32x4 __attribute__((vector_size(4 * sizeof(int))));
typedef __attribute__((address_space(1))) i32x4 as1_i32x4;
typedef __attribute__((address_space(3))) i32x4 as3_i32x4;
static __device__ __forceinline__ void async_copy16(const bf16_t* g, bf16_t* l) {
  __builtin_amdgcn_global_load_async_to_lds_b128(
      (as1_i32x4*)(void*)g, (as3_i32x4*)(void*)l, 0, 0);
}
#if __has_builtin(__builtin_amdgcn_s_wait_asynccnt)
#define WAIT_ASYNC(n) __builtin_amdgcn_s_wait_asynccnt(n)
#else
#define WAIT_ASYNC(n) asm volatile("s_wait_asynccnt " #n ::: "memory")
#endif
#else
#define HAVE_ASYNC_LDS 0
#endif

static __device__ __forceinline__ f32x8 zero8() {
  f32x8 v;
#pragma unroll
  for (int i = 0; i < 8; ++i) v[i] = 0.0f;
  return v;
}

// A fragment (16x32 bf16, M x K). lane<16: row M=lane, K={0..7,16..23};
// lane>=16: row M=lane-16, K={8..15,24..31}.
static __device__ __forceinline__ bf16x16 load_a_frag(const bf16_t* p, int ld, int lane) {
  int lo = lane & 15, hi = (lane >> 4) & 1;
  const bf16_t* q = p + (size_t)lo * ld + hi * 8;
  bf16x8 c0 = *(const bf16x8*)(q);
  bf16x8 c1 = *(const bf16x8*)(q + 16);
  bf16x16 r;
#pragma unroll
  for (int i = 0; i < 8; ++i) { r[i] = c0[i]; r[i + 8] = c1[i]; }
  return r;
}

// B fragment (32x16 bf16, K x N) with memory holding W as [N, K] row-major.
// lane<16 -> N=lane, K={0..7,16..23}; lane>=16 -> N=lane-16, K={8..15,24..31}.
static __device__ __forceinline__ bf16x16 load_b_frag(const bf16_t* p, int ld, int lane) {
  int lo = lane & 15, hi = (lane >> 4) & 1;
  const bf16_t* q = p + (size_t)lo * ld + hi * 8;
  bf16x8 c0 = *(const bf16x8*)(q);
  bf16x8 c1 = *(const bf16x8*)(q + 16);
  bf16x16 r;
#pragma unroll
  for (int i = 0; i < 8; ++i) { r[i] = c0[i]; r[i + 8] = c1[i]; }
  return r;
}

static __device__ __forceinline__ f32x8 wmma_bf16(bf16x16 a, bf16x16 b, f32x8 c) {
  return __builtin_amdgcn_wmma_f32_16x16x32_bf16(false, a, false, b, (short)0, c,
                                                 false, false);
}

// ---------------- fp32 -> bf16 convert ----------------
__global__ void cvt_f32_bf16_kernel(const float* __restrict__ src,
                                    bf16_t* __restrict__ dst, int n) {
  int i = blockIdx.x * blockDim.x + threadIdx.x;
  if (i < n) dst[i] = (bf16_t)src[i];
}

// ---------------- tiled GEMM: out = act(X[M,K] @ W[N,K]^T + bias[N]) --------
// 256 threads = 8 waves. Workgroup tile 128(M) x 128(N), BK = 32.
// Wave (wm, wn) grid 4x2: each wave owns 32(M) x 64(N) = 8 accumulators.
// LDS double-buffered; tile k+1 staged via async global->LDS DMA (ASYNCcnt)
// while WMMAs of tile k run out of LDS.
#define BM 128
#define BN 128
#define BKC 32

__global__ __launch_bounds__(256) void gemm_tiled_kernel(
    const bf16_t* __restrict__ X, const bf16_t* __restrict__ W,
    const float* __restrict__ bias, float* __restrict__ out32,
    bf16_t* __restrict__ out16, int Mr, int Nr, int Kr, int do_relu) {
  __shared__ alignas(16) bf16_t Xs[2][BM][BKC];  // 2 x 8 KB
  __shared__ alignas(16) bf16_t Ws[2][BN][BKC];  // 2 x 8 KB

  int tid = threadIdx.x;
  int lane = tid & 31, wave = tid >> 5;
  int wm = (wave >> 1) * 32;  // 0,32,64,96
  int wn = (wave & 1) * 64;   // 0,64

  int tilesN = Nr / BN;
  int tM = blockIdx.x / tilesN;
  int tN = blockIdx.x - tM * tilesN;
  int M0 = tM * BM, N0 = tN * BN;

  const bf16_t* Xg = X + (size_t)M0 * Kr;
  const bf16_t* Wg = W + (size_t)N0 * Kr;

  f32x8 acc[2][4];
#pragma unroll
  for (int mi = 0; mi < 2; ++mi)
#pragma unroll
    for (int j = 0; j < 4; ++j) acc[mi][j] = zero8();

  int KT = Kr / BKC;

#if HAVE_ASYNC_LDS
  // 128x32 tile = 512 chunks of 16B; 256 threads issue 2 async copies each.
  auto tile_async = [&](const bf16_t* src, int kofs, bf16_t(*dst)[BKC]) {
    int r0 = tid >> 2, c0 = (tid & 3) * 8;
    int r1 = (tid + 256) >> 2, c1 = ((tid + 256) & 3) * 8;
    async_copy16(src + (size_t)r0 * Kr + kofs + c0, &dst[r0][c0]);
    async_copy16(src + (size_t)r1 * Kr + kofs + c1, &dst[r1][c1]);
  };

  tile_async(Xg, 0, Xs[0]);
  tile_async(Wg, 0, Ws[0]);

  for (int kt = 0; kt < KT; ++kt) {
    int buf = kt & 1;
    if (kt + 1 < KT) {  // issue tile k+1 into the other buffer (4 in flight)
      int ko = (kt + 1) * BKC;
      tile_async(Xg, ko, Xs[buf ^ 1]);
      tile_async(Wg, ko, Ws[buf ^ 1]);
      WAIT_ASYNC(4);    // tile k landed; tile k+1 still streaming
    } else {
      WAIT_ASYNC(0);
    }
    __syncthreads();

    bf16x16 a0 = load_a_frag(&Xs[buf][wm][0], BKC, lane);
    bf16x16 a1 = load_a_frag(&Xs[buf][wm + 16][0], BKC, lane);
#pragma unroll
    for (int j = 0; j < 4; ++j) {
      bf16x16 b = load_b_frag(&Ws[buf][wn + j * 16][0], BKC, lane);
      acc[0][j] = wmma_bf16(a0, b, acc[0][j]);
      acc[1][j] = wmma_bf16(a1, b, acc[1][j]);
    }
    __syncthreads();
  }
#else
  // fallback: register-staged double buffering
  auto ldg = [&](const bf16_t* src, int kofs, int c) -> bf16x8 {
    int row = c >> 2, col = (c & 3) * 8;
    return *(const bf16x8*)(src + (size_t)row * Kr + kofs + col);
  };
  auto sts = [&](bf16_t(*dst)[BKC], int c, bf16x8 v) {
    int row = c >> 2, col = (c & 3) * 8;
    *(bf16x8*)(&dst[row][col]) = v;
  };

  bf16x8 rx0 = ldg(Xg, 0, tid), rx1 = ldg(Xg, 0, tid + 256);
  bf16x8 rw0 = ldg(Wg, 0, tid), rw1 = ldg(Wg, 0, tid + 256);
  sts(Xs[0], tid, rx0); sts(Xs[0], tid + 256, rx1);
  sts(Ws[0], tid, rw0); sts(Ws[0], tid + 256, rw1);
  __syncthreads();

  for (int kt = 0; kt < KT; ++kt) {
    int buf = kt & 1;
    if (kt + 1 < KT) {
      int ko = (kt + 1) * BKC;
      rx0 = ldg(Xg, ko, tid); rx1 = ldg(Xg, ko, tid + 256);
      rw0 = ldg(Wg, ko, tid); rw1 = ldg(Wg, ko, tid + 256);
      if (kt + 2 < KT) {
        int kp = (kt + 2) * BKC;
        __builtin_prefetch(Xg + (size_t)(tid >> 1) * Kr + kp, 0, 3);
        __builtin_prefetch(Wg + (size_t)(tid >> 1) * Kr + kp, 0, 3);
      }
    }

    bf16x16 a0 = load_a_frag(&Xs[buf][wm][0], BKC, lane);
    bf16x16 a1 = load_a_frag(&Xs[buf][wm + 16][0], BKC, lane);
#pragma unroll
    for (int j = 0; j < 4; ++j) {
      bf16x16 b = load_b_frag(&Ws[buf][wn + j * 16][0], BKC, lane);
      acc[0][j] = wmma_bf16(a0, b, acc[0][j]);
      acc[1][j] = wmma_bf16(a1, b, acc[1][j]);
    }

    if (kt + 1 < KT) {
      int nb = buf ^ 1;
      sts(Xs[nb], tid, rx0); sts(Xs[nb], tid + 256, rx1);
      sts(Ws[nb], tid, rw0); sts(Ws[nb], tid + 256, rw1);
    }
    __syncthreads();
  }
#endif

  int lo = lane & 15, hi = (lane >> 4) & 1;
#pragma unroll
  for (int mi = 0; mi < 2; ++mi) {
#pragma unroll
    for (int j = 0; j < 4; ++j) {
      int n = N0 + wn + j * 16 + lo;
      float bvv = bias[n];
#pragma unroll
      for (int r = 0; r < 8; ++r) {
        int m = M0 + wm + mi * 16 + r + 8 * hi;
        float v = acc[mi][j][r] + bvv;
        if (do_relu) v = fmaxf(v, 0.0f);
        size_t o = (size_t)m * Nr + n;
        if (out32) out32[o] = v;
        if (out16) out16[o] = (bf16_t)v;
      }
    }
  }
}

// ---------------- fused attention ----------------
// grid: b*H*32 + h*32 + qt  (qt = 16-query tile), 128 threads (4 waves).
__global__ __launch_bounds__(128) void attn_kernel(
    const bf16_t* __restrict__ q16, const bf16_t* __restrict__ k16,
    const bf16_t* __restrict__ v16, float* __restrict__ attn32) {
  __shared__ alignas(16) float  sc[16][SS];   // 32 KB score tile (f32)
  __shared__ alignas(16) bf16_t pm[16][SS];   // 16 KB probabilities (bf16)
  __shared__ alignas(16) bf16_t vT[DKK][64];  // 8 KB transposed V chunk
  __shared__ float rtmp[16][8];

  int idx = blockIdx.x;
  int qt = idx & 31;
  int h  = (idx >> 5) & 7;
  int b  = idx >> 8;
  int tid = threadIdx.x;
  int lane = tid & 31, wave = tid >> 5;
  int lo = lane & 15, hi = (lane >> 4) & 1;

  const bf16_t* Qp = q16 + ((size_t)(b * SS + qt * 16)) * DD + h * DKK;

  // ---- phase 1: S = Q K^T / sqrt(DK);  wave w covers keys [w*128, w*128+128)
  for (int nt = 0; nt < 8; ++nt) {
    int kb = wave * 128 + nt * 16;
    f32x8 acc = zero8();
#pragma unroll
    for (int kk = 0; kk < DKK; kk += 32) {
      bf16x16 a  = load_a_frag(Qp + kk, DD, lane);
      bf16x16 bf = load_b_frag(k16 + ((size_t)(b * SS + kb)) * DD + h * DKK + kk,
                               DD, lane);
      acc = wmma_bf16(a, bf, acc);
    }
#pragma unroll
    for (int r = 0; r < 8; ++r)
      sc[r + 8 * hi][kb + lo] = acc[r] * 0.125f;  // 1/sqrt(64)
  }
  __syncthreads();

  // ---- phase 2: row softmax (8 threads per row, 64 cols each)
  {
    int rr = tid >> 3, seg = tid & 7;
    int c0 = seg * 64;
    float mx = -1e30f;
    for (int k2 = c0; k2 < c0 + 64; ++k2) mx = fmaxf(mx, sc[rr][k2]);
    rtmp[rr][seg] = mx;
    __syncthreads();
    mx = rtmp[rr][0];
#pragma unroll
    for (int j = 1; j < 8; ++j) mx = fmaxf(mx, rtmp[rr][j]);
    float sum = 0.0f;
    for (int k2 = c0; k2 < c0 + 64; ++k2) {
      float e = __expf(sc[rr][k2] - mx);
      sc[rr][k2] = e;
      sum += e;
    }
    __syncthreads();
    rtmp[rr][seg] = sum;
    __syncthreads();
    float tot = 0.0f;
#pragma unroll
    for (int j = 0; j < 8; ++j) tot += rtmp[rr][j];
    float inv = 1.0f / tot;
    for (int k2 = c0; k2 < c0 + 64; ++k2) pm[rr][k2] = (bf16_t)(sc[rr][k2] * inv);
  }
  __syncthreads();

  // ---- phase 3: out = P V; wave w owns dk columns [w*16, w*16+16)
  f32x8 acc = zero8();
  for (int c = 0; c < 8; ++c) {  // key chunks of 64
    __syncthreads();             // previous chunk fully consumed
    for (int t = tid; t < DKK * 64; t += 128) {
      int dk = t >> 6, key = t & 63;
      vT[dk][key] = v16[((size_t)(b * SS + c * 64 + key)) * DD + h * DKK + dk];
    }
    __syncthreads();
#pragma unroll
    for (int kk = 0; kk < 64; kk += 32) {
      bf16x16 a  = load_a_frag(&pm[0][c * 64 + kk], SS, lane);
      bf16x16 bf = load_b_frag(&vT[wave * 16][kk], 64, lane);
      acc = wmma_bf16(a, bf, acc);
    }
  }
#pragma unroll
  for (int r = 0; r < 8; ++r) {
    attn32[((size_t)(b * SS + qt * 16 + r + 8 * hi)) * DD + h * DKK + wave * 16 + lo] =
        acc[r];
  }
}

// ---------------- residual add + LayerNorm (unbiased std, /(std+eps)) --------
__global__ __launch_bounds__(256) void add_ln_kernel(
    const float* __restrict__ A, const float* __restrict__ Bv,
    const float* __restrict__ gamma, const float* __restrict__ beta,
    float* __restrict__ o32, bf16_t* __restrict__ o16) {
  int row = blockIdx.x, tid = threadIdx.x;
  const float* pa = A  + (size_t)row * DD;
  const float* pb = Bv + (size_t)row * DD;
  float v0 = pa[tid] + pb[tid];
  float v1 = pa[tid + 256] + pb[tid + 256];

  __shared__ float wsum[8];
  int lane = tid & 31, wv = tid >> 5;

  float s = v0 + v1;
#pragma unroll
  for (int o = 16; o > 0; o >>= 1) s += __shfl_xor(s, o, 32);
  if (lane == 0) wsum[wv] = s;
  __syncthreads();
  float tot = 0.0f;
#pragma unroll
  for (int j = 0; j < 8; ++j) tot += wsum[j];
  float mean = tot * (1.0f / 512.0f);

  float d0 = v0 - mean, d1 = v1 - mean;
  float s2 = d0 * d0 + d1 * d1;
#pragma unroll
  for (int o = 16; o > 0; o >>= 1) s2 += __shfl_xor(s2, o, 32);
  __syncthreads();
  if (lane == 0) wsum[wv] = s2;
  __syncthreads();
  float tot2 = 0.0f;
#pragma unroll
  for (int j = 0; j < 8; ++j) tot2 += wsum[j];
  float var = tot2 * (1.0f / 511.0f);        // ddof=1
  float inv = 1.0f / (sqrtf(var) + 1e-8f);   // gamma*(x-mean)/(std+eps)+beta

  float r0 = gamma[tid] * d0 * inv + beta[tid];
  float r1 = gamma[tid + 256] * d1 * inv + beta[tid + 256];
  size_t o0 = (size_t)row * DD + tid, o1 = o0 + 256;
  if (o32) { o32[o0] = r0; o32[o1] = r1; }
  if (o16) { o16[o0] = (bf16_t)r0; o16[o1] = (bf16_t)r1; }
}

// ---------------- host-side orchestration ----------------
extern "C" void kernel_launch(void* const* d_in, const int* in_sizes, int n_in,
                              void* d_out, int out_size, void* d_ws, size_t ws_size,
                              hipStream_t stream) {
  (void)in_sizes; (void)n_in; (void)out_size; (void)ws_size;
  const float* x      = (const float*)d_in[0];
  const float* wq     = (const float*)d_in[1];
  const float* bq     = (const float*)d_in[2];
  const float* wk     = (const float*)d_in[3];
  const float* bk     = (const float*)d_in[4];
  const float* wv     = (const float*)d_in[5];
  const float* bv     = (const float*)d_in[6];
  const float* wr     = (const float*)d_in[7];
  const float* br     = (const float*)d_in[8];
  const float* g_attn = (const float*)d_in[9];
  const float* b_attn = (const float*)d_in[10];
  const float* wc1    = (const float*)d_in[11];
  const float* bc1    = (const float*)d_in[12];
  const float* wc2    = (const float*)d_in[13];
  const float* bc2    = (const float*)d_in[14];
  const float* g_ffn  = (const float*)d_in[15];
  const float* b_ffn  = (const float*)d_in[16];

  // carve workspace
  char* p = (char*)d_ws;
  auto carve = [&](size_t bytes) -> void* {
    void* r = (void*)p;
    p += (bytes + 255) & ~(size_t)255;
    return r;
  };
  const size_t Wdd = (size_t)NBB * DD * DD;
  const size_t Wdf = (size_t)NBB * DFF * DD;
  const size_t MD  = (size_t)MROWS * DD;
  const size_t MDF = (size_t)MROWS * DFF;

  bf16_t* wq16  = (bf16_t*)carve(Wdd * 2);
  bf16_t* wk16  = (bf16_t*)carve(Wdd * 2);
  bf16_t* wv16  = (bf16_t*)carve(Wdd * 2);
  bf16_t* wr16  = (bf16_t*)carve(Wdd * 2);
  bf16_t* wc116 = (bf16_t*)carve(Wdf * 2);
  bf16_t* wc216 = (bf16_t*)carve(Wdf * 2);
  bf16_t* x16   = (bf16_t*)carve(MD * 2);
  bf16_t* q16   = (bf16_t*)carve(MD * 2);
  bf16_t* k16   = (bf16_t*)carve(MD * 2);
  bf16_t* v16   = (bf16_t*)carve(MD * 2);
  float*  r32   = (float*)carve(MD * 4);
  float*  at32  = (float*)carve(MD * 4);
  float*  ln132 = (float*)carve(MD * 4);
  bf16_t* ln116 = (bf16_t*)carve(MD * 2);
  bf16_t* h16   = (bf16_t*)carve(MDF * 2);
  float*  y32   = (float*)carve(MD * 4);
  float*  enc32 = (float*)carve(MD * 4);

  auto cvt = [&](const float* src, bf16_t* dst, size_t n) {
    cvt_f32_bf16_kernel<<<(unsigned)((n + 255) / 256), 256, 0, stream>>>(src, dst, (int)n);
  };
  auto gemm = [&](const bf16_t* Xp, const bf16_t* Wp, const float* bias,
                  float* o32, bf16_t* o16, int Mr, int Nr, int Kr, int relu) {
    int grid = (Mr / BM) * (Nr / BN);
    gemm_tiled_kernel<<<grid, 256, 0, stream>>>(Xp, Wp, bias, o32, o16, Mr, Nr, Kr, relu);
  };

  // one-time weight conversion (all blocks)
  cvt(wq, wq16, Wdd);  cvt(wk, wk16, Wdd);
  cvt(wv, wv16, Wdd);  cvt(wr, wr16, Wdd);
  cvt(wc1, wc116, Wdf); cvt(wc2, wc216, Wdf);

  const float* enc_in = x;
  for (int i = 0; i < NBB; ++i) {
    cvt(enc_in, x16, MD);

    gemm(x16, wq16 + (size_t)i * DD * DD, bq + i * DD, nullptr, q16, MROWS, DD, DD, 1);
    gemm(x16, wk16 + (size_t)i * DD * DD, bk + i * DD, nullptr, k16, MROWS, DD, DD, 1);
    gemm(x16, wv16 + (size_t)i * DD * DD, bv + i * DD, nullptr, v16, MROWS, DD, DD, 1);
    gemm(x16, wr16 + (size_t)i * DD * DD, br + i * DD, r32, nullptr, MROWS, DD, DD, 1);

    attn_kernel<<<BB * HH * (SS / 16), 128, 0, stream>>>(q16, k16, v16, at32);

    add_ln_kernel<<<MROWS, 256, 0, stream>>>(at32, r32, g_attn + i * DD,
                                             b_attn + i * DD, ln132, ln116);

    gemm(ln116, wc116 + (size_t)i * DFF * DD, bc1 + i * DFF, nullptr, h16,
         MROWS, DFF, DD, 1);
    gemm(h16, wc216 + (size_t)i * DD * DFF, bc2 + i * DD, y32, nullptr,
         MROWS, DD, DFF, 0);

    float* dst = (i == NBB - 1) ? (float*)d_out : enc32;
    add_ln_kernel<<<MROWS, 256, 0, stream>>>(y32, ln132, g_ffn + i * DD,
                                             b_ffn + i * DD, dst, nullptr);
    enc_in = enc32;
  }
}
